// SAMBA_17514876633685
// MI455X (gfx1250) — matrix operand order repaired and checked by reading, not compiled, not run on hardware
//
#include <hip/hip_runtime.h>
#include <hip/hip_bf16.h>
#include <math.h>

// SAMBA forward for MI455X (gfx1250).
// - All GEMMs via v_wmma_f32_16x16x32_f16: f16 activations, weights
//   pre-transposed+converted to f16 (N x K) so every B fragment is one
//   aligned 32B vector load (two global_load_b128).
// - Block tile 32x256: each wave computes a 32x64 strip = 8 WMMAs per
//   k-step (two A fragments share four B fragments; reuse_a hints set).
// - The block-shared 32x32 A tile is staged to LDS with the gfx1250 async
//   global->LDS path (double buffered, s_wait_asynccnt), with a synchronous
//   fallback if the builtins are unavailable.
// - Selective scan: one lane per (d, n) state, 16-lane shfl-xor reduction.

#define N_LAYERS  4
#define D_MODEL   512
#define D_INNER   1024
#define D_STATE   16
#define DT_RANK   32
#define SEQ_LEN   1024
#define HID_DIM   2048
#define BATCH_N   4
#define ROWS      (BATCH_N * SEQ_LEN)   // 4096
#define EPSV      1e-5f

typedef __attribute__((ext_vector_type(16))) _Float16     v16h;
typedef __attribute__((ext_vector_type(8)))  float        v8f;
typedef __attribute__((ext_vector_type(4)))  int          v4i;
typedef __attribute__((ext_vector_type(8)))  unsigned int v8u;

enum { ACT_NONE = 0, ACT_RELU = 1, ACT_SOFTPLUS = 2 };

#if defined(__HIP_DEVICE_COMPILE__) && \
    __has_builtin(__builtin_amdgcn_global_load_async_to_lds_b128) && \
    __has_builtin(__builtin_amdgcn_s_wait_asynccnt)
#define USE_ASYNC_LDS 1
#else
#define USE_ASYNC_LDS 0
#endif

__device__ __forceinline__ void stage_tile(const _Float16* g, _Float16* l)
{
#if USE_ASYNC_LDS
    __builtin_amdgcn_global_load_async_to_lds_b128(
        (__attribute__((address_space(1))) v4i*)(v4i*)(void*)const_cast<_Float16*>(g),
        (__attribute__((address_space(3))) v4i*)(v4i*)(void*)l, 0, 0);
#else
    *(uint4*)(void*)l = *(const uint4*)(const void*)g;
#endif
}

__device__ __forceinline__ void stage_wait()
{
#if USE_ASYNC_LDS
    __builtin_amdgcn_s_wait_asynccnt(0);
#endif
    __syncthreads();
}

struct U8x32 { uint4 a, b; };   // 32-byte carrier for a v16h fragment

__device__ __forceinline__ v16h load_fragB(const _Float16* p)   // 32B-aligned
{
    return __builtin_bit_cast(v16h, *(const v8u*)p);
}

__device__ __forceinline__ v16h load_fragA_lds(const _Float16* p)  // 16B-aligned
{
    U8x32 r;
    r.a = *(const uint4*)p;
    r.b = *(const uint4*)(p + 8);
    return __builtin_bit_cast(v16h, r);
}

// ---------------------------------------------------------------------------
// WMMA GEMM: C[M,N] = act(A[M,K](f16) * Bt[N,K](f16)^T + bias)
// grid = (M/32, ceil(N/256)); block = 128 (4 waves); wave strip = 32x64.
// A tile (32x32 halfs) staged to LDS (async, double buffered), shared by all
// 4 waves. A/B fragments use contiguous k order (k = 16*(lane/16) + s).
// ---------------------------------------------------------------------------
template <typename OutT>
__global__ __launch_bounds__(128)
void gemm_wmma2(const _Float16* __restrict__ A, int lda,
                const _Float16* __restrict__ Bt,      // N x K, f16
                const float* __restrict__ bias,
                OutT* __restrict__ C, int ldc,
                int M, int N, int K, int act)
{
    __shared__ _Float16 sA[2][32 * 32];

    const int lane  = threadIdx.x & 31;
    const int wave  = threadIdx.x >> 5;
    const int tileM = blockIdx.x << 5;            // 32 rows per block

    int tn = (blockIdx.y * 4 + wave) << 6;        // 64 columns per wave
    const bool active = (tn < N);
    if (!active) tn = (N > 64) ? (N - 64) : 0;    // clamp: compute in-bounds, skip store

    const int rowA = lane & 15;
    const int khiA = (lane >> 4) << 4;            // 0 or 16 (halfs)
    const int khiB = (lane >> 4) << 4;            // 0 or 16 (halfs)
    const int colB = lane & 15;

    const _Float16* Brow0 = Bt + (size_t)(tn +  0 + colB) * K + khiB;
    const _Float16* Brow1 = Bt + (size_t)(tn + 16 + colB) * K + khiB;
    const _Float16* Brow2 = Bt + (size_t)(tn + 32 + colB) * K + khiB;
    const _Float16* Brow3 = Bt + (size_t)(tn + 48 + colB) * K + khiB;

    v8f a0t0 = {}, a0t1 = {}, a0t2 = {}, a0t3 = {};   // rows tileM..+15
    v8f a1t0 = {}, a1t1 = {}, a1t2 = {}, a1t3 = {};   // rows tileM+16..+31

    const int srow = threadIdx.x >> 2;            // 0..31, all 128 threads stage
    const int sch  = (threadIdx.x & 3) << 3;      // 4 x 8-half chunks per row
    stage_tile(A + (size_t)(tileM + srow) * lda + sch, &sA[0][srow * 32 + sch]);
    stage_wait();

    int buf = 0;
    for (int k0 = 0; k0 < K; k0 += 32) {
        if (k0 + 32 < K) {
            stage_tile(A + (size_t)(tileM + srow) * lda + (k0 + 32) + sch,
                       &sA[buf ^ 1][srow * 32 + sch]);
            __builtin_prefetch(Brow0 + k0 + 32, 0, 0);   // global_prefetch_b8
            __builtin_prefetch(Brow2 + k0 + 32, 0, 0);
        }

        const v16h af0 = load_fragA_lds(&sA[buf][(rowA     ) * 32 + khiA]);
        const v16h af1 = load_fragA_lds(&sA[buf][(rowA + 16) * 32 + khiA]);

        const v16h b0 = load_fragB(Brow0 + k0);
        const v16h b1 = load_fragB(Brow1 + k0);
        const v16h b2 = load_fragB(Brow2 + k0);
        const v16h b3 = load_fragB(Brow3 + k0);

        a0t0 = __builtin_amdgcn_wmma_f32_16x16x32_f16(false, af0, false, b0, (short)0, a0t0, false, false);
        a0t1 = __builtin_amdgcn_wmma_f32_16x16x32_f16(false, af0, false, b1, (short)0, a0t1, true,  false);
        a0t2 = __builtin_amdgcn_wmma_f32_16x16x32_f16(false, af0, false, b2, (short)0, a0t2, true,  false);
        a0t3 = __builtin_amdgcn_wmma_f32_16x16x32_f16(false, af0, false, b3, (short)0, a0t3, true,  false);
        a1t0 = __builtin_amdgcn_wmma_f32_16x16x32_f16(false, af1, false, b0, (short)0, a1t0, false, false);
        a1t1 = __builtin_amdgcn_wmma_f32_16x16x32_f16(false, af1, false, b1, (short)0, a1t1, true,  false);
        a1t2 = __builtin_amdgcn_wmma_f32_16x16x32_f16(false, af1, false, b2, (short)0, a1t2, true,  false);
        a1t3 = __builtin_amdgcn_wmma_f32_16x16x32_f16(false, af1, false, b3, (short)0, a1t3, true,  false);

        stage_wait();
        buf ^= 1;
    }

    if (active) {
        const int rhalf = (lane >> 4) << 3;       // 0 or 8
        auto epi = [&](const v8f& a, int mo, int t) {
            const int cn = tn + (t << 4) + colB;
            const int rbase = tileM + mo + rhalf;
            const float bv = bias ? bias[cn] : 0.0f;
#pragma unroll
            for (int j = 0; j < 8; ++j) {
                float v = a[j] + bv;
                if (act == ACT_RELU)          v = fmaxf(v, 0.0f);
                else if (act == ACT_SOFTPLUS) v = (v > 20.0f) ? v : log1pf(__expf(v));
                C[(size_t)(rbase + j) * ldc + cn] = (OutT)v;
            }
        };
        epi(a0t0,  0, 0); epi(a0t1,  0, 1); epi(a0t2,  0, 2); epi(a0t3,  0, 3);
        epi(a1t0, 16, 0); epi(a1t1, 16, 1); epi(a1t2, 16, 2); epi(a1t3, 16, 3);
    }
}

// W (K x N, f32) -> Wt (N x K, f16), 32x32 LDS tile transpose.
__global__ __launch_bounds__(256)
void transpose_f2h(const float* __restrict__ W, _Float16* __restrict__ Wt,
                   int K, int N)
{
    __shared__ float tile[32][33];
    const int tx = threadIdx.x & 31, ty = threadIdx.x >> 5;   // 32 x 8
    const int n0 = blockIdx.x << 5, k0 = blockIdx.y << 5;
    for (int r = ty; r < 32; r += 8) {
        const int k = k0 + r, n = n0 + tx;
        tile[r][tx] = (k < K && n < N) ? W[(size_t)k * N + n] : 0.0f;
    }
    __syncthreads();
    for (int r = ty; r < 32; r += 8) {
        const int n = n0 + r, k = k0 + tx;
        if (n < N && k < K) Wt[(size_t)n * K + k] = (_Float16)tile[tx][r];
    }
}

// ---------------------------------------------------------------------------
// LayerNorm / RMSNorm across last dim, optional seq flip on the read side.
// ---------------------------------------------------------------------------
__global__ __launch_bounds__(256)
void norm_rows(const float* __restrict__ x, const float* __restrict__ w,
               const float* __restrict__ b, _Float16* __restrict__ y,
               int rowlen, int rms, int flip)
{
    const int r = blockIdx.x;
    int srcr = r;
    if (flip) {
        const int bb = r / SEQ_LEN, l = r % SEQ_LEN;
        srcr = bb * SEQ_LEN + (SEQ_LEN - 1 - l);
    }
    const float* xr = x + (size_t)srcr * rowlen;

    float s = 0.f, s2 = 0.f;
    for (int i = threadIdx.x; i < rowlen; i += blockDim.x) {
        const float v = xr[i]; s += v; s2 += v * v;
    }
    for (int m = 16; m >= 1; m >>= 1) {
        s  += __shfl_xor(s,  m, 32);
        s2 += __shfl_xor(s2, m, 32);
    }
    __shared__ float sh[2][8];
    const int wv = threadIdx.x >> 5;
    if ((threadIdx.x & 31) == 0) { sh[0][wv] = s; sh[1][wv] = s2; }
    __syncthreads();
    if (threadIdx.x == 0) {
        float a = 0.f, c = 0.f;
        for (int i = 0; i < (int)(blockDim.x >> 5); ++i) { a += sh[0][i]; c += sh[1][i]; }
        sh[0][0] = a; sh[1][0] = c;
    }
    __syncthreads();
    const float inv  = 1.0f / (float)rowlen;
    const float mean = rms ? 0.0f : sh[0][0] * inv;
    const float var  = sh[1][0] * inv - mean * mean;
    const float sc   = rsqrtf(var + EPSV);
    for (int i = threadIdx.x; i < rowlen; i += blockDim.x) {
        const float v = (xr[i] - mean) * sc * w[i] + (rms ? 0.0f : b[i]);
        y[(size_t)r * rowlen + i] = (_Float16)v;
    }
}

// Norm over the SEQ axis of x(b,l,d) for fixed (b,d) -> y[(b*D+d), l] (f16).
__global__ __launch_bounds__(256)
void norm_seq(const float* __restrict__ x, const float* __restrict__ w,
              const float* __restrict__ b, _Float16* __restrict__ y, int rms)
{
    const int r  = blockIdx.x;               // b*D_MODEL + d
    const int bb = r / D_MODEL, d = r % D_MODEL;
    const float* base = x + (size_t)bb * SEQ_LEN * D_MODEL + d;

    float s = 0.f, s2 = 0.f;
    for (int l = threadIdx.x; l < SEQ_LEN; l += blockDim.x) {
        const float v = base[(size_t)l * D_MODEL]; s += v; s2 += v * v;
    }
    for (int m = 16; m >= 1; m >>= 1) {
        s  += __shfl_xor(s,  m, 32);
        s2 += __shfl_xor(s2, m, 32);
    }
    __shared__ float sh[2][8];
    const int wv = threadIdx.x >> 5;
    if ((threadIdx.x & 31) == 0) { sh[0][wv] = s; sh[1][wv] = s2; }
    __syncthreads();
    if (threadIdx.x == 0) {
        float a = 0.f, c = 0.f;
        for (int i = 0; i < (int)(blockDim.x >> 5); ++i) { a += sh[0][i]; c += sh[1][i]; }
        sh[0][0] = a; sh[1][0] = c;
    }
    __syncthreads();
    const float inv  = 1.0f / (float)SEQ_LEN;
    const float mean = rms ? 0.0f : sh[0][0] * inv;
    const float var  = sh[1][0] * inv - mean * mean;
    const float sc   = rsqrtf(var + EPSV);
    for (int l = threadIdx.x; l < SEQ_LEN; l += blockDim.x) {
        const float v = (base[(size_t)l * D_MODEL] - mean) * sc * w[l]
                        + (rms ? 0.0f : b[l]);
        y[(size_t)r * SEQ_LEN + l] = (_Float16)v;
    }
}

// Depthwise causal conv (K=3, left pad 2) + SiLU. xs = first half of xr rows.
__global__ __launch_bounds__(256)
void conv_silu(const float* __restrict__ xr, const float* __restrict__ cw,
               const float* __restrict__ cb, _Float16* __restrict__ xc16,
               float* __restrict__ xc32)
{
    const size_t i = (size_t)blockIdx.x * blockDim.x + threadIdx.x;
    if (i >= (size_t)ROWS * D_INNER) return;
    const int    d  = (int)(i % D_INNER);
    const size_t bl = i / D_INNER;
    const int    l  = (int)(bl % SEQ_LEN);
    const int    bb = (int)(bl / SEQ_LEN);
    const float* xs = xr + (size_t)bb * SEQ_LEN * (2 * D_INNER) + d;
    float acc = cb[d];
#pragma unroll
    for (int k = 0; k < 3; ++k) {
        const int ls = l - 2 + k;
        if (ls >= 0) acc += xs[(size_t)ls * (2 * D_INNER)] * cw[d * 3 + k];
    }
    const float sv = acc / (1.0f + __expf(-acc));
    xc32[i] = sv;
    xc16[i] = (_Float16)sv;
}

// Selective scan: one lane per (d, n); 16-lane reduction over n.
__global__ __launch_bounds__(256)
void scan_kernel(const float* __restrict__ delta,  // (ROWS, D_INNER)
                 const float* __restrict__ u,      // (ROWS, D_INNER)
                 const float* __restrict__ xdbl,   // (ROWS, 64): B @ 32+n, C @ 48+n
                 const float* __restrict__ Alog,   // (D_INNER, 16)
                 const float* __restrict__ Dp,     // (D_INNER)
                 float* __restrict__ y)            // (ROWS, D_INNER)
{
    const int gid = blockIdx.x * blockDim.x + threadIdx.x;
    const int n   = gid & 15;
    const int ch  = gid >> 4;                // [0, BATCH_N*D_INNER)
    const int bb  = ch / D_INNER;
    const int d   = ch % D_INNER;

    const float A  = -__expf(Alog[d * D_STATE + n]);
    const float Dd = Dp[d];

    const float* drow = delta + (size_t)bb * SEQ_LEN * D_INNER + d;
    const float* urow = u     + (size_t)bb * SEQ_LEN * D_INNER + d;
    const float* xrow = xdbl  + (size_t)bb * SEQ_LEN * 64;
    float*       yrow = y     + (size_t)bb * SEQ_LEN * D_INNER + d;

    float s = 0.0f;
    for (int l = 0; l < SEQ_LEN; ++l) {
        const float dt = drow[(size_t)l * D_INNER];
        const float ut = urow[(size_t)l * D_INNER];
        const float Bt = xrow[(size_t)l * 64 + DT_RANK + n];
        const float Ct = xrow[(size_t)l * 64 + DT_RANK + D_STATE + n];
        s = __expf(dt * A) * s + dt * ut * Bt;
        float p = s * Ct;
        p += __shfl_xor(p, 1, 32);
        p += __shfl_xor(p, 2, 32);
        p += __shfl_xor(p, 4, 32);
        p += __shfl_xor(p, 8, 32);
        if (n == 0) yrow[(size_t)l * D_INNER] = p + ut * Dd;
    }
}

// z16 = (f16)(y * silu(res)); res lives in second half of xr rows.
__global__ __launch_bounds__(256)
void gate_silu(const float* __restrict__ y, const float* __restrict__ xr,
               _Float16* __restrict__ z)
{
    const size_t i = (size_t)blockIdx.x * blockDim.x + threadIdx.x;
    if (i >= (size_t)ROWS * D_INNER) return;
    const int    d  = (int)(i % D_INNER);
    const size_t bl = i / D_INNER;
    const float  r  = xr[bl * (2 * D_INNER) + D_INNER + d];
    const float  g  = r / (1.0f + __expf(-r));
    z[i] = (_Float16)(y[i] * g);
}

__global__ __launch_bounds__(256)
void cvt_f2h(const float* __restrict__ x, _Float16* __restrict__ y, int n)
{
    const int i = blockIdx.x * blockDim.x + threadIdx.x;
    if (i < n) y[i] = (_Float16)x[i];
}

// delta16[r, 0:32] = (f16) xdbl[r, 0:32]
__global__ __launch_bounds__(256)
void cvt_delta(const float* __restrict__ xdbl, _Float16* __restrict__ d16)
{
    const int i = blockIdx.x * blockDim.x + threadIdx.x;
    if (i >= ROWS * DT_RANK) return;
    const int r = i / DT_RANK, c = i % DT_RANK;
    d16[i] = (_Float16)xdbl[(size_t)r * 64 + c];
}

// x = y1 + flip(y2, seq) + x
__global__ __launch_bounds__(256)
void combine(float* __restrict__ x, const float* __restrict__ y1,
             const float* __restrict__ y2)
{
    const size_t i = (size_t)blockIdx.x * blockDim.x + threadIdx.x;
    if (i >= (size_t)ROWS * D_MODEL) return;
    const int    d  = (int)(i % D_MODEL);
    const size_t bl = i / D_MODEL;
    const int    l  = (int)(bl % SEQ_LEN);
    const int    bb = (int)(bl / SEQ_LEN);
    const size_t j  = ((size_t)bb * SEQ_LEN + (SEQ_LEN - 1 - l)) * D_MODEL + d;
    x[i] = y1[i] + y2[j] + x[i];
}

// x[b,l,d] += h2[(b*D_MODEL+d), l]  (MLP residual, transposed)
__global__ __launch_bounds__(256)
void add_tr(float* __restrict__ x, const float* __restrict__ h2)
{
    const size_t i = (size_t)blockIdx.x * blockDim.x + threadIdx.x;
    if (i >= (size_t)ROWS * D_MODEL) return;
    const int    d  = (int)(i % D_MODEL);
    const size_t bl = i / D_MODEL;
    const int    l  = (int)(bl % SEQ_LEN);
    const int    bb = (int)(bl / SEQ_LEN);
    x[i] += h2[((size_t)bb * D_MODEL + d) * SEQ_LEN + l];
}

// ---------------------------------------------------------------------------
extern "C" void kernel_launch(void* const* d_in, const int* in_sizes, int n_in,
                              void* d_out, int out_size, void* d_ws, size_t ws_size,
                              hipStream_t stream)
{
    (void)in_sizes; (void)n_in; (void)out_size; (void)ws_size;

    const float* inp     = (const float*)d_in[0];
    const float* W_emb   = (const float*)d_in[1];
    const float* b_emb   = (const float*)d_in[2];
    const float* ln_w[2] = {(const float*)d_in[3],  (const float*)d_in[14]};
    const float* ln_b[2] = {(const float*)d_in[4],  (const float*)d_in[15]};
    const float* Win[2]  = {(const float*)d_in[5],  (const float*)d_in[16]};
    const float* cwp[2]  = {(const float*)d_in[6],  (const float*)d_in[17]};
    const float* cbp[2]  = {(const float*)d_in[7],  (const float*)d_in[18]};
    const float* Wx[2]   = {(const float*)d_in[8],  (const float*)d_in[19]};
    const float* Wdt[2]  = {(const float*)d_in[9],  (const float*)d_in[20]};
    const float* bdt[2]  = {(const float*)d_in[10], (const float*)d_in[21]};
    const float* Alog[2] = {(const float*)d_in[11], (const float*)d_in[22]};
    const float* Dp[2]   = {(const float*)d_in[12], (const float*)d_in[23]};
    const float* Wout[2] = {(const float*)d_in[13], (const float*)d_in[24]};
    const float* lnl_w   = (const float*)d_in[25];
    const float* lnl_b   = (const float*)d_in[26];
    const float* Wl1     = (const float*)d_in[27];
    const float* bl1     = (const float*)d_in[28];
    const float* Wl2     = (const float*)d_in[29];
    const float* bl2     = (const float*)d_in[30];
    const float* normf_w = (const float*)d_in[31];
    const float* normf_b = (const float*)d_in[32];
    const float* W_head  = (const float*)d_in[33];
    const float* b_head  = (const float*)d_in[34];

    // ---- workspace carve (256B aligned) ----
    char* wsb = (char*)d_ws;
    size_t off = 0;
    auto carve = [&](size_t bytes) -> void* {
        void* r = wsb + off;
        off = (off + bytes + 255) & ~(size_t)255;
        return r;
    };
    float*    X    = (float*)   carve((size_t)ROWS * D_MODEL * 4);
    float*    Y1   = (float*)   carve((size_t)ROWS * D_MODEL * 4);
    float*    Y2   = (float*)   carve((size_t)ROWS * D_MODEL * 4);
    float*    XR   = (float*)   carve((size_t)ROWS * 2 * D_INNER * 4);
    _Float16* A16  = (_Float16*)carve((size_t)ROWS * D_INNER * 4);  // reused as f32 y
    _Float16* B16  = (_Float16*)carve((size_t)ROWS * D_INNER * 2);
    float*    F1   = (float*)   carve((size_t)ROWS * D_INNER * 4);  // xc32 / h2
    float*    F2   = (float*)   carve((size_t)ROWS * D_INNER * 4);  // delta
    float*    XDBL = (float*)   carve((size_t)ROWS * 64 * 4);
    _Float16* D16  = (_Float16*)carve((size_t)ROWS * DT_RANK * 2);
    float*    YF32 = (float*)A16;

    // transposed f16 weights (N x K)
    _Float16* tWemb = (_Float16*)carve((size_t)D_MODEL * 128 * 2);
    _Float16 *tWin[2][N_LAYERS], *tWx[2][N_LAYERS], *tWdt[2][N_LAYERS], *tWout[2][N_LAYERS];
    _Float16 *tWl1[N_LAYERS], *tWl2[N_LAYERS];
    for (int d = 0; d < 2; ++d)
        for (int L = 0; L < N_LAYERS; ++L) {
            tWin[d][L]  = (_Float16*)carve((size_t)2 * D_INNER * D_MODEL * 2);
            tWx[d][L]   = (_Float16*)carve((size_t)(DT_RANK + 2 * D_STATE) * D_INNER * 2);
            tWdt[d][L]  = (_Float16*)carve((size_t)D_INNER * DT_RANK * 2);
            tWout[d][L] = (_Float16*)carve((size_t)D_MODEL * D_INNER * 2);
        }
    for (int L = 0; L < N_LAYERS; ++L) {
        tWl1[L] = (_Float16*)carve((size_t)HID_DIM * SEQ_LEN * 2);
        tWl2[L] = (_Float16*)carve((size_t)SEQ_LEN * HID_DIM * 2);
    }
    _Float16* tWhead = (_Float16*)carve((size_t)128 * D_MODEL * 2);

    auto tr = [&](const float* W, _Float16* Wt, int K, int N) {
        transpose_f2h<<<dim3((N + 31) / 32, (K + 31) / 32), 256, 0, stream>>>(W, Wt, K, N);
    };
    tr(W_emb, tWemb, 128, D_MODEL);
    for (int d = 0; d < 2; ++d)
        for (int L = 0; L < N_LAYERS; ++L) {
            tr(Win[d]  + (size_t)L * D_MODEL * 2 * D_INNER, tWin[d][L],  D_MODEL, 2 * D_INNER);
            tr(Wx[d]   + (size_t)L * D_INNER * (DT_RANK + 2 * D_STATE), tWx[d][L], D_INNER, DT_RANK + 2 * D_STATE);
            tr(Wdt[d]  + (size_t)L * DT_RANK * D_INNER, tWdt[d][L], DT_RANK, D_INNER);
            tr(Wout[d] + (size_t)L * D_INNER * D_MODEL, tWout[d][L], D_INNER, D_MODEL);
        }
    for (int L = 0; L < N_LAYERS; ++L) {
        tr(Wl1 + (size_t)L * SEQ_LEN * HID_DIM, tWl1[L], SEQ_LEN, HID_DIM);
        tr(Wl2 + (size_t)L * HID_DIM * SEQ_LEN, tWl2[L], HID_DIM, SEQ_LEN);
    }
    tr(W_head, tWhead, D_MODEL, 128);

    auto gemmF = [&](const _Float16* A, int lda, const _Float16* Bt,
                     const float* bias, float* C, int ldc,
                     int M, int N, int K, int act) {
        gemm_wmma2<float><<<dim3(M / 32, (N + 255) / 256), 128, 0, stream>>>(
            A, lda, Bt, bias, C, ldc, M, N, K, act);
    };
    auto gemmH = [&](const _Float16* A, int lda, const _Float16* Bt,
                     const float* bias, _Float16* C, int ldc,
                     int M, int N, int K, int act) {
        gemm_wmma2<_Float16><<<dim3(M / 32, (N + 255) / 256), 128, 0, stream>>>(
            A, lda, Bt, bias, C, ldc, M, N, K, act);
    };

    const int NELEM_DI = ROWS * D_INNER;   // 4 M
    const int NELEM_DM = ROWS * D_MODEL;   // 2 M

    // ---- embedding: X = inp @ W_emb + b_emb ----
    cvt_f2h<<<(ROWS * 128 + 255) / 256, 256, 0, stream>>>(inp, A16, ROWS * 128);
    gemmF(A16, 128, tWemb, b_emb, X, D_MODEL, ROWS, D_MODEL, 128, ACT_NONE);

    for (int L = 0; L < N_LAYERS; ++L) {
        for (int dir = 0; dir < 2; ++dir) {
            const float* lw   = ln_w[dir] + L * D_MODEL;
            const float* lb   = ln_b[dir] + L * D_MODEL;
            const float* cw   = cwp[dir]  + (size_t)L * D_INNER * 3;
            const float* cb   = cbp[dir]  + (size_t)L * D_INNER;
            const float* bdtp = bdt[dir]  + (size_t)L * D_INNER;
            const float* al   = Alog[dir] + (size_t)L * D_INNER * D_STATE;
            const float* dpp  = Dp[dir]   + (size_t)L * D_INNER;
            float* Y = dir ? Y2 : Y1;

            // norm (+ seq flip for dir 1) -> f16
            norm_rows<<<ROWS, 256, 0, stream>>>(X, lw, lb, A16, D_MODEL, 0, dir);
            // xr = xln @ W_in (4096 x 2048)
            gemmF(A16, D_MODEL, tWin[dir][L], nullptr, XR, 2 * D_INNER,
                  ROWS, 2 * D_INNER, D_MODEL, ACT_NONE);
            // depthwise causal conv + SiLU -> xc16 / xc32
            conv_silu<<<(NELEM_DI + 255) / 256, 256, 0, stream>>>(XR, cw, cb, B16, F1);
            // x_dbl = xc @ W_x (4096 x 64)
            gemmF(B16, D_INNER, tWx[dir][L], nullptr, XDBL, DT_RANK + 2 * D_STATE,
                  ROWS, DT_RANK + 2 * D_STATE, D_INNER, ACT_NONE);
            // delta = softplus(x_dbl[:, :32] @ W_dt + b_dt)
            cvt_delta<<<(ROWS * DT_RANK + 255) / 256, 256, 0, stream>>>(XDBL, D16);
            gemmF(D16, DT_RANK, tWdt[dir][L], bdtp, F2, D_INNER,
                  ROWS, D_INNER, DT_RANK, ACT_SOFTPLUS);
            // selective scan
            scan_kernel<<<(BATCH_N * D_INNER * D_STATE) / 256, 256, 0, stream>>>(
                F2, F1, XDBL, al, dpp, YF32);
            // z = y * silu(res) -> f16
            gate_silu<<<(NELEM_DI + 255) / 256, 256, 0, stream>>>(YF32, XR, B16);
            // Y = z @ W_out
            gemmF(B16, D_INNER, tWout[dir][L], nullptr, Y, D_MODEL,
                  ROWS, D_MODEL, D_INNER, ACT_NONE);
        }
        // x = y1 + flip(y2) + x
        combine<<<(NELEM_DM + 255) / 256, 256, 0, stream>>>(X, Y1, Y2);

        // ---- seq-axis MLP on x^T ----
        norm_seq<<<BATCH_N * D_MODEL, 256, 0, stream>>>(
            X, lnl_w + L * SEQ_LEN, lnl_b + L * SEQ_LEN, A16, (L == 0) ? 0 : 1);
        gemmH(A16, SEQ_LEN, tWl1[L], bl1 + (size_t)L * HID_DIM, B16, HID_DIM,
              BATCH_N * D_MODEL, HID_DIM, SEQ_LEN, ACT_RELU);
        gemmF(B16, HID_DIM, tWl2[L], bl2 + (size_t)L * SEQ_LEN, F1, SEQ_LEN,
              BATCH_N * D_MODEL, SEQ_LEN, HID_DIM, ACT_NONE);
        add_tr<<<(NELEM_DM + 255) / 256, 256, 0, stream>>>(X, F1);
    }

    // ---- final norm + head ----
    norm_rows<<<ROWS, 256, 0, stream>>>(X, normf_w, normf_b, A16, D_MODEL, 0, 0);
    gemmF(A16, D_MODEL, tWhead, b_head, (float*)d_out, 128,
          ROWS, 128, D_MODEL, ACT_NONE);
}
